// EasyPreTrainLatentAttentionWithoutIndexer_69681549410473
// MI455X (gfx1250) — matrix operand order repaired and checked by reading, not compiled, not run on hardware
//
#include <hip/hip_runtime.h>

// ---------------- problem constants ----------------
#define DIMX    2048
#define SEQ     2048
#define BATCH   2
#define NH      16
#define Q_LORA  1536
#define KV_LORA 512
#define NOPE    128
#define ROPEDIM 64
#define VHD     128
#define QK_HD   192           // NOPE + ROPE
#define TOKENS  (BATCH*SEQ)   // 4096

typedef __bf16 bf16;
typedef __bf16 v8bf  __attribute__((ext_vector_type(8)));
typedef __bf16 v16bf __attribute__((ext_vector_type(16)));
typedef float  v8f   __attribute__((ext_vector_type(8)));

// ---------------- WMMA fragment helpers (wave32, 16x16x32 bf16) ----------------
// A (16x32, M x K): lane half=lane>>4, m=lane&15.
//   vector elems 0..7  = A[m][k0 + half*8 .. +7]
//   vector elems 8..15 = A[m][k0 + 16 + half*8 .. +7]
__device__ __forceinline__ v16bf load_frag_a(const bf16* p, int ld, int lane) {
  int half = lane >> 4, m = lane & 15;
  const bf16* r = p + (size_t)m * ld + half * 8;
  union { v16bf v; v8bf h[2]; } u;
  u.h[0] = *(const v8bf*)(r);
  u.h[1] = *(const v8bf*)(r + 16);
  return u.v;
}
// B (32x16, K x N) given as row-major W[N][K] (column n of B == row n of W):
//   lane n=lane&15, half=lane>>4 holds W[n][k0 + half*8 .. +7], W[n][k0 + 16 + half*8 .. +7]
__device__ __forceinline__ v16bf load_frag_b(const bf16* p, int ld, int lane) {
  int half = lane >> 4, n = lane & 15;
  const bf16* r = p + (size_t)n * ld + half * 8;
  union { v16bf v; v8bf h[2]; } u;
  u.h[0] = *(const v8bf*)(r);
  u.h[1] = *(const v8bf*)(r + 16);
  return u.v;
}
__device__ __forceinline__ v8f wmma_bf16(v16bf a, v16bf b, v8f c) {
  return __builtin_amdgcn_wmma_f32_16x16x32_bf16(false, a, false, b, (short)0, c, false, false);
}

// ---------------- async cache->LDS copy (gfx1250 GLOBAL_LOAD_ASYNC_TO_LDS) ------
// Generic pointer to LDS: low 32 bits are the wave-relative LDS byte offset,
// which is what the instruction expects as its LDS address operand.
__device__ __forceinline__ void async_copy_b128(const bf16* gsrc, bf16* ldst) {
  unsigned loff = (unsigned)(unsigned long long)(void*)ldst;
  asm volatile("global_load_async_to_lds_b128 %0, %1, off"
               :: "v"(loff), "v"(gsrc) : "memory");
}
__device__ __forceinline__ void async_wait0() {
  asm volatile("s_wait_asynccnt 0x0" ::: "memory");
}

// ---------------- fp32 -> bf16 convert ----------------
__global__ void cvt_f32_bf16(const float* __restrict__ in, bf16* __restrict__ out, int n) {
  int i = blockIdx.x * 256 + threadIdx.x;
  if (i < n) out[i] = (bf16)in[i];
}

// ---------------- GEMM: C[M,N] = A_bf16[M,K] * W_bf16[N,K]^T + bias[N] (fp32 out) ----
// block = 256 threads = 8 waves, block tile 256(M) x 64(N), wave tile 32x64.
// 6 fragment loads -> 8 WMMAs per wave K-step; weight columns shared by all
// 8 waves stay hot in L0/L2 (all bf16 weights are L2-resident on MI455X).
__global__ __launch_bounds__(256) void gemm_bf16_f32(
    const bf16* __restrict__ A, const bf16* __restrict__ W,
    const float* __restrict__ bias, float* __restrict__ C,
    int M, int N, int K) {
  const int w = threadIdx.x >> 5, lane = threadIdx.x & 31;
  const int wm = blockIdx.y * 256 + w * 32;
  const int wn = blockIdx.x * 64;
  v8f a00 = {}, a01 = {}, a02 = {}, a03 = {};
  v8f a10 = {}, a11 = {}, a12 = {}, a13 = {};
  const bf16* a0p = A + (size_t)wm * K;
  const bf16* a1p = A + (size_t)(wm + 16) * K;
  const bf16* b0p = W + (size_t)(wn + 0) * K;
  const bf16* b1p = W + (size_t)(wn + 16) * K;
  const bf16* b2p = W + (size_t)(wn + 32) * K;
  const bf16* b3p = W + (size_t)(wn + 48) * K;
  for (int k = 0; k < K; k += 32) {
    if (k + 256 < K) {
      __builtin_prefetch((const void*)(a0p + k + 256), 0, 1);
      __builtin_prefetch((const void*)(b0p + k + 256), 0, 1);
    }
    v16bf fa0 = load_frag_a(a0p + k, K, lane);
    v16bf fa1 = load_frag_a(a1p + k, K, lane);
    v16bf fb0 = load_frag_b(b0p + k, K, lane);
    v16bf fb1 = load_frag_b(b1p + k, K, lane);
    v16bf fb2 = load_frag_b(b2p + k, K, lane);
    v16bf fb3 = load_frag_b(b3p + k, K, lane);
    a00 = wmma_bf16(fa0, fb0, a00);
    a01 = wmma_bf16(fa0, fb1, a01);
    a02 = wmma_bf16(fa0, fb2, a02);
    a03 = wmma_bf16(fa0, fb3, a03);
    a10 = wmma_bf16(fa1, fb0, a10);
    a11 = wmma_bf16(fa1, fb1, a11);
    a12 = wmma_bf16(fa1, fb2, a12);
    a13 = wmma_bf16(fa1, fb3, a13);
  }
  const int half = lane >> 4, n = lane & 15;
  auto st = [&](const v8f& a, int r0, int c0) {
    float bv = bias[c0 + n];
#pragma unroll
    for (int v = 0; v < 8; v++)
      C[(size_t)(r0 + half * 8 + v) * N + c0 + n] = a[v] + bv;
  };
  st(a00, wm, wn);      st(a01, wm, wn + 16);      st(a02, wm, wn + 32);      st(a03, wm, wn + 48);
  st(a10, wm + 16, wn); st(a11, wm + 16, wn + 16); st(a12, wm + 16, wn + 32); st(a13, wm + 16, wn + 48);
}

// ---------------- RMSNorm: per row, fp32 in (strided) -> bf16 out ----------------
__global__ __launch_bounds__(256) void rmsnorm_bf16(
    const float* __restrict__ in, int in_stride,
    const float* __restrict__ wt, bf16* __restrict__ out, int C) {
  const float* x = in + (size_t)blockIdx.x * in_stride;
  bf16* o = out + (size_t)blockIdx.x * C;
  __shared__ float red[256];
  float s = 0.f;
  for (int i = threadIdx.x; i < C; i += 256) { float v = x[i]; s += v * v; }
  red[threadIdx.x] = s;
  __syncthreads();
  for (int off = 128; off > 0; off >>= 1) {
    if ((int)threadIdx.x < off) red[threadIdx.x] += red[threadIdx.x + off];
    __syncthreads();
  }
  float r = rsqrtf(red[0] / (float)C + 1.1920929e-07f);
  for (int i = threadIdx.x; i < C; i += 256) o[i] = (bf16)(x[i] * r * wt[i]);
}

// ---------------- Q: rope + fold 1/sqrt(192) scale; (B,S,H,192)f32 -> (B,H,S,192)bf16
__global__ void rope_scale_q(const float* __restrict__ Qpre,
                             const float* __restrict__ cosb, const float* __restrict__ sinb,
                             bf16* __restrict__ Qh) {
  size_t idx = (size_t)blockIdx.x * 256 + threadIdx.x;
  const size_t total = (size_t)BATCH * NH * SEQ * QK_HD;
  if (idx >= total) return;
  int d = (int)(idx % QK_HD); size_t r = idx / QK_HD;
  int s = (int)(r % SEQ); size_t r2 = r / SEQ;
  int h = (int)(r2 % NH); int b = (int)(r2 / NH);
  const float* src = Qpre + ((size_t)(b * SEQ + s) * NH + h) * QK_HD;
  float val;
  if (d < NOPE) {
    val = src[d];
  } else {
    int t = (d - NOPE) >> 1;
    float c = cosb[s * 32 + t], sn = sinb[s * 32 + t];
    float xr = src[NOPE + 2 * t], xi = src[NOPE + 2 * t + 1];
    val = ((d & 1) == 0) ? (xr * c - xi * sn) : (xr * sn + xi * c);
  }
  Qh[idx] = (bf16)(val * 0.07216878364870322f);  // 192^-0.5
}

// ---------------- k_rope: kv_pre[...,512:576] -> rope -> (B,S,64) bf16 ----------------
__global__ void rope_k(const float* __restrict__ kvpre,
                       const float* __restrict__ cosb, const float* __restrict__ sinb,
                       bf16* __restrict__ kro) {
  int idx = blockIdx.x * 256 + threadIdx.x;
  if (idx >= TOKENS * ROPEDIM) return;
  int d = idx % ROPEDIM, tok = idx / ROPEDIM, s = tok % SEQ;
  const float* src = kvpre + (size_t)tok * (KV_LORA + ROPEDIM) + KV_LORA;
  int t = d >> 1;
  float c = cosb[s * 32 + t], sn = sinb[s * 32 + t];
  float xr = src[2 * t], xi = src[2 * t + 1];
  kro[idx] = (bf16)(((d & 1) == 0) ? (xr * c - xi * sn) : (xr * sn + xi * c));
}

// ---------------- K build: knv nope + broadcast k_rope -> (B,H,S,192) bf16 ----------------
__global__ void build_k(const float* __restrict__ knv, const bf16* __restrict__ kro,
                        bf16* __restrict__ Kh) {
  size_t idx = (size_t)blockIdx.x * 256 + threadIdx.x;
  const size_t total = (size_t)BATCH * NH * SEQ * QK_HD;
  if (idx >= total) return;
  int d = (int)(idx % QK_HD); size_t r = idx / QK_HD;
  int s = (int)(r % SEQ); size_t r2 = r / SEQ;
  int h = (int)(r2 % NH); int b = (int)(r2 / NH);
  size_t tok = (size_t)b * SEQ + s;
  bf16 val;
  if (d < NOPE) val = (bf16)knv[(tok * NH + h) * (NOPE + VHD) + d];
  else          val = kro[tok * ROPEDIM + (d - NOPE)];
  Kh[idx] = val;
}

// ---------------- V transpose: knv[...,128:256] -> (B,H,128,S) bf16 ----------------
__global__ void build_vt(const float* __restrict__ knv, bf16* __restrict__ Vt) {
  size_t idx = (size_t)blockIdx.x * 256 + threadIdx.x;
  const size_t total = (size_t)BATCH * NH * VHD * SEQ;
  if (idx >= total) return;
  int s = (int)(idx % SEQ); size_t r = idx / SEQ;
  int d = (int)(r % VHD); size_t r2 = r / VHD;
  int h = (int)(r2 % NH); int b = (int)(r2 / NH);
  size_t tok = (size_t)b * SEQ + s;
  Vt[idx] = (bf16)knv[(tok * NH + h) * (NOPE + VHD) + NOPE + d];
}

// ---------------- flash attention (causal), WMMA bf16, async LDS staging ----------------
#define TQ 64
#define TK 32
__global__ __launch_bounds__(256) void flash_attn(
    const bf16* __restrict__ Qh, const bf16* __restrict__ Kh,
    const bf16* __restrict__ Vt, bf16* __restrict__ Out) {
  extern __shared__ char smem[];
  bf16* sQ  = (bf16*)smem;           // 64*192
  bf16* sK  = sQ + TQ * QK_HD;       // 32*192
  bf16* sVt = sK + TK * QK_HD;       // 128*32 (vdim-major)
  bf16* sP  = sVt + VHD * TK;        // 64*32
  float* sS = (float*)(sP + TQ * TK);// 64*32
  float* sM = sS + TQ * TK;
  float* sL = sM + TQ;
  float* sA = sL + TQ;

  const int b = blockIdx.z, h = blockIdx.y, q0 = blockIdx.x * TQ;
  const bf16* Qbh = Qh + ((size_t)(b * NH + h)) * SEQ * QK_HD + (size_t)q0 * QK_HD;
  const bf16* Kbh = Kh + ((size_t)(b * NH + h)) * SEQ * QK_HD;
  const bf16* Vbh = Vt + ((size_t)(b * NH + h)) * VHD * SEQ;

  const int tid = threadIdx.x, w = tid >> 5, lane = tid & 31;
  const int half = lane >> 4;

  // Q tile: global -> LDS via async DMA (1536 x b128, exact 6 per thread)
  for (int i = tid; i < TQ * QK_HD / 8; i += 256)
    async_copy_b128(Qbh + (size_t)i * 8, sQ + (size_t)i * 8);
  if (tid < TQ) { sM[tid] = -1e30f; sL[tid] = 0.f; }
  async_wait0();
  __syncthreads();

  const int m0 = (w & 3) * 16;   // wave's query-row tile base
  const int ng = (w >> 2) * 4;   // wave's first of 4 vdim tiles (of 8)
  v8f o0 = {}, o1 = {}, o2 = {}, o3 = {};

  for (int k0 = 0; k0 <= q0 + TQ - TK; k0 += TK) {
    // stage K tile and transposed V tile: async cache->LDS
    for (int i = tid; i < TK * QK_HD / 8; i += 256)     // 768 -> 3 per thread
      async_copy_b128(Kbh + (size_t)k0 * QK_HD + (size_t)i * 8, sK + (size_t)i * 8);
    for (int i = tid; i < VHD * TK / 8; i += 256) {     // 512 -> 2 per thread
      int dd = i / (TK / 8), jj = i % (TK / 8);
      async_copy_b128(Vbh + (size_t)dd * SEQ + k0 + jj * 8, sVt + (size_t)i * 8);
    }
    async_wait0();
    __syncthreads();

    // S = Q * K^T : 4 Mtiles x 2 Ntiles, one 16x16 tile per wave
    {
      int ti = w & 3, tj = w >> 2;
      v8f sacc = {};
#pragma unroll
      for (int kk = 0; kk < QK_HD; kk += 32) {
        v16bf af  = load_frag_a(sQ + ti * 16 * QK_HD + kk, QK_HD, lane);
        v16bf bfg = load_frag_b(sK + tj * 16 * QK_HD + kk, QK_HD, lane);
        sacc = wmma_bf16(af, bfg, sacc);
      }
      int n = lane & 15;
#pragma unroll
      for (int v = 0; v < 8; v++)
        sS[(ti * 16 + half * 8 + v) * TK + tj * 16 + n] = sacc[v];
    }
    __syncthreads();

    // online softmax: one thread per query row
    if (tid < TQ) {
      int r = tid, q = q0 + r;
      float mx = -1e30f;
      for (int j = 0; j < TK; j++)
        if (k0 + j <= q) mx = fmaxf(mx, sS[r * TK + j]);
      float mold = sM[r];
      float mnew = fmaxf(mold, mx);
      float alpha = __expf(mold - mnew);
      float sum = 0.f;
      for (int j = 0; j < TK; j++) {
        float p = (k0 + j <= q) ? __expf(sS[r * TK + j] - mnew) : 0.f;
        sP[r * TK + j] = (bf16)p;
        sum += p;
      }
      sM[r] = mnew; sL[r] = sL[r] * alpha + sum; sA[r] = alpha;
    }
    __syncthreads();

    // rescale O and accumulate P*V
    float ar[8];
#pragma unroll
    for (int v = 0; v < 8; v++) ar[v] = sA[m0 + half * 8 + v];
#pragma unroll
    for (int v = 0; v < 8; v++) {
      o0[v] *= ar[v]; o1[v] *= ar[v]; o2[v] *= ar[v]; o3[v] *= ar[v];
    }
    v16bf pf = load_frag_a(sP + m0 * TK, TK, lane);
    o0 = wmma_bf16(pf, load_frag_b(sVt + (ng + 0) * 16 * TK, TK, lane), o0);
    o1 = wmma_bf16(pf, load_frag_b(sVt + (ng + 1) * 16 * TK, TK, lane), o1);
    o2 = wmma_bf16(pf, load_frag_b(sVt + (ng + 2) * 16 * TK, TK, lane), o2);
    o3 = wmma_bf16(pf, load_frag_b(sVt + (ng + 3) * 16 * TK, TK, lane), o3);
    __syncthreads();
  }

  // normalize by l and write (B,S,H*VHD)
  int n = lane & 15;
  float lr[8];
#pragma unroll
  for (int v = 0; v < 8; v++) lr[v] = 1.f / sL[m0 + half * 8 + v];
  v8f* oacc[4] = {&o0, &o1, &o2, &o3};
#pragma unroll
  for (int j = 0; j < 4; j++) {
    int col = (ng + j) * 16 + n;
#pragma unroll
    for (int v = 0; v < 8; v++) {
      int row = m0 + half * 8 + v;
      Out[((size_t)(b * SEQ + q0 + row)) * (NH * VHD) + h * VHD + col] =
          (bf16)((*oacc[j])[v] * lr[v]);
    }
  }
}

// ---------------- host orchestration ----------------
extern "C" void kernel_launch(void* const* d_in, const int* in_sizes, int n_in,
                              void* d_out, int out_size, void* d_ws, size_t ws_size,
                              hipStream_t stream) {
  (void)in_sizes; (void)n_in; (void)out_size; (void)ws_size;
  const float* x    = (const float*)d_in[0];
  const float* fcos = (const float*)d_in[1];
  const float* fsin = (const float*)d_in[2];
  const float* Wqd  = (const float*)d_in[3];
  const float* bqd  = (const float*)d_in[4];
  const float* qnw  = (const float*)d_in[5];
  const float* Wqu  = (const float*)d_in[6];
  const float* bqu  = (const float*)d_in[7];
  const float* Wkvd = (const float*)d_in[8];
  const float* bkvd = (const float*)d_in[9];
  const float* kvnw = (const float*)d_in[10];
  const float* Wkvu = (const float*)d_in[11];
  const float* bkvu = (const float*)d_in[12];
  const float* Wo   = (const float*)d_in[13];
  const float* bo   = (const float*)d_in[14];

  char* ws = (char*)d_ws;
  size_t off = 0;
  auto alloc = [&](size_t bytes) -> void* {
    void* p = ws + off;
    off += (bytes + 255) & ~(size_t)255;
    return p;
  };

  bf16* xb      = (bf16*)alloc((size_t)TOKENS * DIMX * 2);
  bf16* Wqd_b   = (bf16*)alloc((size_t)Q_LORA * DIMX * 2);
  bf16* Wqu_b   = (bf16*)alloc((size_t)(NH * QK_HD) * Q_LORA * 2);
  bf16* Wkvd_b  = (bf16*)alloc((size_t)(KV_LORA + ROPEDIM) * DIMX * 2);
  bf16* Wkvu_b  = (bf16*)alloc((size_t)(NH * (NOPE + VHD)) * KV_LORA * 2);
  bf16* Wo_b    = (bf16*)alloc((size_t)DIMX * (NH * VHD) * 2);
  float* qlatp  = (float*)alloc((size_t)TOKENS * Q_LORA * 4);
  bf16* qlat    = (bf16*)alloc((size_t)TOKENS * Q_LORA * 2);
  float* Qpre   = (float*)alloc((size_t)TOKENS * NH * QK_HD * 4);
  bf16* QhB     = (bf16*)alloc((size_t)TOKENS * NH * QK_HD * 2);
  float* kvpre  = (float*)alloc((size_t)TOKENS * (KV_LORA + ROPEDIM) * 4);
  bf16* kvlat   = (bf16*)alloc((size_t)TOKENS * KV_LORA * 2);
  bf16* kro     = (bf16*)alloc((size_t)TOKENS * ROPEDIM * 2);
  float* knv    = (float*)alloc((size_t)TOKENS * NH * (NOPE + VHD) * 4);
  bf16* KhB     = (bf16*)alloc((size_t)TOKENS * NH * QK_HD * 2);
  bf16* VtB     = (bf16*)alloc((size_t)TOKENS * NH * VHD * 2);
  bf16* attnO   = (bf16*)alloc((size_t)TOKENS * NH * VHD * 2);

  auto cvt = [&](const float* in, bf16* out, size_t n) {
    cvt_f32_bf16<<<dim3((unsigned)((n + 255) / 256)), dim3(256), 0, stream>>>(in, out, (int)n);
  };
  auto gemm = [&](const bf16* A, const bf16* W, const float* bias, float* C,
                  int M, int N, int K) {
    gemm_bf16_f32<<<dim3(N / 64, M / 256), dim3(256), 0, stream>>>(A, W, bias, C, M, N, K);
  };

  // bf16 conversions (weights fit in 192MB L2 afterwards)
  cvt(x, xb, (size_t)TOKENS * DIMX);
  cvt(Wqd, Wqd_b, (size_t)Q_LORA * DIMX);
  cvt(Wqu, Wqu_b, (size_t)(NH * QK_HD) * Q_LORA);
  cvt(Wkvd, Wkvd_b, (size_t)(KV_LORA + ROPEDIM) * DIMX);
  cvt(Wkvu, Wkvu_b, (size_t)(NH * (NOPE + VHD)) * KV_LORA);
  cvt(Wo, Wo_b, (size_t)DIMX * (NH * VHD));

  // q path
  gemm(xb, Wqd_b, bqd, qlatp, TOKENS, Q_LORA, DIMX);
  rmsnorm_bf16<<<dim3(TOKENS), dim3(256), 0, stream>>>(qlatp, Q_LORA, qnw, qlat, Q_LORA);
  gemm(qlat, Wqu_b, bqu, Qpre, TOKENS, NH * QK_HD, Q_LORA);
  {
    size_t total = (size_t)BATCH * NH * SEQ * QK_HD;
    rope_scale_q<<<dim3((unsigned)((total + 255) / 256)), dim3(256), 0, stream>>>(
        Qpre, fcos, fsin, QhB);
  }

  // kv path
  gemm(xb, Wkvd_b, bkvd, kvpre, TOKENS, KV_LORA + ROPEDIM, DIMX);
  rmsnorm_bf16<<<dim3(TOKENS), dim3(256), 0, stream>>>(kvpre, KV_LORA + ROPEDIM, kvnw, kvlat, KV_LORA);
  rope_k<<<dim3((TOKENS * ROPEDIM + 255) / 256), dim3(256), 0, stream>>>(kvpre, fcos, fsin, kro);
  gemm(kvlat, Wkvu_b, bkvu, knv, TOKENS, NH * (NOPE + VHD), KV_LORA);
  {
    size_t totK = (size_t)BATCH * NH * SEQ * QK_HD;
    build_k<<<dim3((unsigned)((totK + 255) / 256)), dim3(256), 0, stream>>>(knv, kro, KhB);
    size_t totV = (size_t)BATCH * NH * VHD * SEQ;
    build_vt<<<dim3((unsigned)((totV + 255) / 256)), dim3(256), 0, stream>>>(knv, VtB);
  }

  // attention
  {
    size_t smem = (size_t)TQ * QK_HD * 2 + (size_t)TK * QK_HD * 2 + (size_t)VHD * TK * 2 +
                  (size_t)TQ * TK * 2 + (size_t)TQ * TK * 4 + (size_t)TQ * 3 * 4;
    flash_attn<<<dim3(SEQ / TQ, NH, BATCH), dim3(256), smem, stream>>>(QhB, KhB, VtB, attnO);
  }

  // output projection -> d_out (fp32)
  gemm(attnO, Wo_b, bo, (float*)d_out, TOKENS, DIMX, NH * VHD);
}
